// MHSAModule_41858751267211
// MI455X (gfx1250) — matrix operand order repaired and checked by reading, not compiled, not run on hardware
//
#include <hip/hip_runtime.h>
#include <hip/hip_bf16.h>

// ---------------- problem constants ----------------
#define BB 4
#define TT 1024
#define DD 512
#define HH 8
#define KK 64
#define LL 2047       // 2T-1 relative positions
#define LP 2048       // padded rows per batch for p (band loads touch row 2047)
#define HK 512        // H*K
#define LN_EPS 1e-3f

typedef __bf16 bf16;
typedef __attribute__((ext_vector_type(16))) __bf16 v16bf;
typedef __attribute__((ext_vector_type(8)))  float  v8f;

struct alignas(16) bf16x8 { bf16 v[8]; };

// D = A(16x32 bf16) * B(32x16 bf16) + C(16x16 f32)
__device__ inline v8f wmma_bf16(v16bf a, v16bf b, v8f c) {
  return __builtin_amdgcn_wmma_f32_16x16x32_bf16(false, a, false, b, (short)0, c,
                                                 false, false);
}

__device__ inline v16bf frag_from_chunks(bf16x8 c0, bf16x8 c1) {
  v16bf f;
#pragma unroll
  for (int e = 0; e < 8; ++e) { f[e] = c0.v[e]; f[e + 8] = c1.v[e]; }
  return f;
}

// A-operand fragment (16 rows x 32 K). Caller supplies this lane's row pointer
// (row = tile_row + (lane&15)). Per ISA: half h reads K-cols [8h..8h+7] and
// [16+8h..16+8h+7] as two 16-byte chunks.
__device__ inline v16bf load_frag_a(const bf16* rowptr, int lane) {
  int h = lane >> 4;
  bf16x8 c0 = *(const bf16x8*)(rowptr + 8 * h);
  bf16x8 c1 = *(const bf16x8*)(rowptr + 16 + 8 * h);
  return frag_from_chunks(c0, c1);
}

// B-operand fragment for B = X^T, with X stored row-major (16 rows N x 32 cols K).
// lane holds col n = lane&15; half h reads K [8h..8h+7] and [16+8h..16+8h+7].
__device__ inline v16bf load_frag_b(const bf16* baseT, int stride, int lane) {
  const bf16* rp = baseT + (size_t)(lane & 15) * stride + 8 * (lane >> 4);
  bf16x8 c0 = *(const bf16x8*)rp;
  bf16x8 c1 = *(const bf16x8*)(rp + 16);
  return frag_from_chunks(c0, c1);
}

// ---------------- LayerNorm (pre-norm) -> bf16 ----------------
__global__ __launch_bounds__(256, 1)
void ln_kernel(const float* __restrict__ x, const float* __restrict__ g,
               const float* __restrict__ be, bf16* __restrict__ y) {
  int row = blockIdx.x;                       // B*T rows
  const float* xr = x + (size_t)row * DD;
  float v0 = xr[threadIdx.x], v1 = xr[threadIdx.x + 256];
  float s = v0 + v1, sq = v0 * v0 + v1 * v1;
#pragma unroll
  for (int m = 1; m < 32; m <<= 1) {
    s += __shfl_xor(s, m, 32);
    sq += __shfl_xor(sq, m, 32);
  }
  __shared__ float ss[8], ssq[8];
  int w = threadIdx.x >> 5;
  if ((threadIdx.x & 31) == 0) { ss[w] = s; ssq[w] = sq; }
  __syncthreads();
  float ts = 0.f, tsq = 0.f;
#pragma unroll
  for (int i = 0; i < 8; ++i) { ts += ss[i]; tsq += ssq[i]; }
  float mu = ts * (1.0f / DD);
  float var = tsq * (1.0f / DD) - mu * mu;
  float rs = rsqrtf(var + LN_EPS);
  bf16* yr = y + (size_t)row * DD;
  yr[threadIdx.x]       = (bf16)((v0 - mu) * rs * g[threadIdx.x] + be[threadIdx.x]);
  yr[threadIdx.x + 256] = (bf16)((v1 - mu) * rs * g[threadIdx.x + 256] + be[threadIdx.x + 256]);
}

// ---------------- weight transpose + bf16 convert ----------------
__global__ __launch_bounds__(256, 1)
void prep_w(const float* __restrict__ Wq, const float* __restrict__ Wk,
            const float* __restrict__ Wv, const float* __restrict__ Wp,
            const float* __restrict__ Wo,
            bf16* __restrict__ WqT, bf16* __restrict__ WkT,
            bf16* __restrict__ WvT, bf16* __restrict__ WpT,
            bf16* __restrict__ WoT) {
  int i = blockIdx.x * 256 + threadIdx.x;     // 512*512 elements
  int r = i >> 9, c = i & 511;
  int ti = c * 512 + r;                       // transposed index
  WqT[ti] = (bf16)Wq[i];
  WkT[ti] = (bf16)Wk[i];
  WvT[ti] = (bf16)Wv[i];
  WpT[ti] = (bf16)Wp[i];
  WoT[ti] = (bf16)Wo[i];                      // Wo[hk][d] -> WoT[d][hk]
}

__global__ __launch_bounds__(256, 1)
void cvt_pos(const float* __restrict__ pos, bf16* __restrict__ pb, int n) {
  int i = blockIdx.x * 256 + threadIdx.x;
  if (i < n) pb[i] = (bf16)pos[i];
}

// ---------------- projection GEMMs (wave tile: 32 x 64) ----------------
// Each B-fragment is reused by two A-row-subtiles: 8 WMMAs per 6 fragment loads.
// MODE 0: q -> qu = q+bq+u and qv = q+bq+v (dual bf16 output)
// MODE 1: k -> k+bk (bf16, row-major [B*T, HK])
// MODE 2: v -> v+bv stored transposed vT[b,h,kd,T]
// MODE 3: p (no bias), M = B*LP with per-batch row clamp to L-1
template <int MODE>
__global__ __launch_bounds__(256, 1)
void gemm_proj(const bf16* __restrict__ A, const bf16* __restrict__ WT,
               const float* __restrict__ bias0, const float* __restrict__ bias1,
               const float* __restrict__ bias2,
               bf16* __restrict__ out0, bf16* __restrict__ out1) {
  int lane = threadIdx.x & 31;
  int widx = blockIdx.x * 8 + (threadIdx.x >> 5);
  int mtile = widx >> 3, ntile = widx & 7;
  int n0 = ntile * 64;
  int m0 = mtile * 32;
  const bf16 *arow0, *arow1;
  if (MODE == 3) {
    int r0 = m0 + (lane & 15);
    int b = r0 >> 11;                         // / LP (tile never straddles batches)
    int l0 = r0 & (LP - 1);
    int l1 = l0 + 16;
    if (l0 >= LL) l0 = LL - 1;                // clamp pad rows (never selected later)
    if (l1 >= LL) l1 = LL - 1;
    arow0 = A + ((size_t)b * LL + l0) * DD;
    arow1 = A + ((size_t)b * LL + l1) * DD;
  } else {
    arow0 = A + (size_t)(m0 + (lane & 15)) * DD;
    arow1 = arow0 + (size_t)16 * DD;
  }
  v8f acc[2][4] = {};
  for (int kc = 0; kc < DD; kc += 32) {
    v16bf a0 = load_frag_a(arow0 + kc, lane);
    v16bf a1 = load_frag_a(arow1 + kc, lane);
#pragma unroll
    for (int f = 0; f < 4; ++f) {
      v16bf bf = load_frag_b(WT + (size_t)(n0 + f * 16) * DD + kc, DD, lane);
      acc[0][f] = wmma_bf16(a0, bf, acc[0][f]);
      acc[1][f] = wmma_bf16(a1, bf, acc[1][f]);
    }
  }
  int hl = lane >> 4, nlo = lane & 15;
#pragma unroll
  for (int ms = 0; ms < 2; ++ms) {
#pragma unroll
    for (int f = 0; f < 4; ++f) {
#pragma unroll
      for (int r = 0; r < 8; ++r) {
        int row = m0 + ms * 16 + r + 8 * hl;
        int col = n0 + f * 16 + nlo;
        float val = acc[ms][f][r];
        if (MODE == 0) {
          float base = val + bias0[col];
          out0[(size_t)row * HK + col] = (bf16)(base + bias1[col]);
          out1[(size_t)row * HK + col] = (bf16)(base + bias2[col]);
        } else if (MODE == 1) {
          out0[(size_t)row * HK + col] = (bf16)(val + bias0[col]);
        } else if (MODE == 2) {
          float v = val + bias0[col];
          int b = row >> 10, t = row & (TT - 1);
          int h = col >> 6, kd = col & 63;
          out0[(((size_t)b * HH + h) * KK + kd) * TT + t] = (bf16)v;
        } else {
          out0[(size_t)row * HK + col] = (bf16)val;   // row already in [0, B*LP)
        }
      }
    }
  }
}

// ---------------- flash attention with rel-shift ----------------
// One wave per (b, h, 16-row query block). 32 keys / iteration.
// The 48-wide positional band overlaps between iterations: G2(s0) == G0(s0+32),
// so the leading band fragment is carried across iterations.
__global__ __launch_bounds__(128, 1)
void attn_kernel(const bf16* __restrict__ qu, const bf16* __restrict__ qv,
                 const bf16* __restrict__ kb, const bf16* __restrict__ pb,
                 const bf16* __restrict__ vT, bf16* __restrict__ ctx) {
  __shared__ bf16 lds[4][16 * 40];            // per-wave 16x32 P tile, padded stride 40
  int lane = threadIdx.x & 31;
  int w = threadIdx.x >> 5;
  int wg = blockIdx.x * 4 + w;
  int b = wg >> 9;                            // / (H * T/16) = /512
  int rem = wg & 511;
  int h = rem >> 6;
  int t0 = (rem & 63) << 4;
  bf16* sl = lds[w];
  int hl = lane >> 4, nlo = lane & 15;

  const bf16* qub = qu + (size_t)(b * TT + t0 + nlo) * HK + h * KK;
  const bf16* qvb = qv + (size_t)(b * TT + t0 + nlo) * HK + h * KK;
  v16bf quA0 = load_frag_a(qub, lane), quA1 = load_frag_a(qub + 32, lane);
  v16bf qvA0 = load_frag_a(qvb, lane), qvA1 = load_frag_a(qvb + 32, lane);

  float mrow[8], lrow[8];
#pragma unroll
  for (int r = 0; r < 8; ++r) { mrow[r] = -3.0e38f; lrow[r] = 0.f; }
  v8f O0 = {}, O1 = {}, O2 = {}, O3 = {};

  const float scale = 0.125f;                 // 1/sqrt(64)
  const bf16* kbh = kb + (size_t)b * TT * HK + h * KK;
  const bf16* pbh = pb + (size_t)b * LP * HK + h * KK;
  const bf16* vth = vT + (size_t)(b * HH + h) * KK * TT;

  // prologue: leading band fragment for s0 = 0 (l in [lbase0, lbase0+15])
  int lbase0 = (TT - 16) - t0;                // >= 0
  v8f G0 = {};
  G0 = wmma_bf16(qvA0, load_frag_b(pbh + (size_t)lbase0 * HK, HK, lane), G0);
  G0 = wmma_bf16(qvA1, load_frag_b(pbh + (size_t)lbase0 * HK + 32, HK, lane), G0);

  for (int s0 = 0; s0 < TT; s0 += 32) {
    if (s0 + 32 < TT) {
      __builtin_prefetch(kbh + (size_t)(s0 + 32) * HK, 0, 1);
      __builtin_prefetch(pbh + (size_t)(lbase0 + s0 + 48) * HK, 0, 1);
    }
    // content scores: two 16x16 subtiles, K=64 in two chunks
    v8f S0 = {}, S1 = {};
    S0 = wmma_bf16(quA0, load_frag_b(kbh + (size_t)s0 * HK, HK, lane), S0);
    S0 = wmma_bf16(quA1, load_frag_b(kbh + (size_t)s0 * HK + 32, HK, lane), S0);
    S1 = wmma_bf16(quA0, load_frag_b(kbh + (size_t)(s0 + 16) * HK, HK, lane), S1);
    S1 = wmma_bf16(quA1, load_frag_b(kbh + (size_t)(s0 + 16) * HK + 32, HK, lane), S1);
    // positional band: l in [lbase, lbase+47]; G0 carried from prev iteration
    int lbase = lbase0 + s0;
    v8f G1 = {}, G2 = {};
    G1 = wmma_bf16(qvA0, load_frag_b(pbh + (size_t)(lbase + 16) * HK, HK, lane), G1);
    G1 = wmma_bf16(qvA1, load_frag_b(pbh + (size_t)(lbase + 16) * HK + 32, HK, lane), G1);
    G2 = wmma_bf16(qvA0, load_frag_b(pbh + (size_t)(lbase + 32) * HK, HK, lane), G2);
    G2 = wmma_bf16(qvA1, load_frag_b(pbh + (size_t)(lbase + 32) * HK + 32, HK, lane), G2);

    // rel-shift gather + online softmax, per accumulator row
#pragma unroll
    for (int r = 0; r < 8; ++r) {
      int i = r + 8 * hl;                     // query row in tile
      int c0 = nlo + 15 - i;                  // band col for subtile 0: [0,30]
      int c1 = c0 + 16;                       // subtile 1: [16,46]
      int src0 = (lane & 16) | (c0 & 15);
      int src1 = (lane & 16) | (c1 & 15);
      float g00 = __shfl(G0[r], src0, 32);
      float g01 = __shfl(G1[r], src0, 32);
      float g10 = __shfl(G1[r], src1, 32);
      float g11 = __shfl(G2[r], src1, 32);
      float ga = (c0 < 16) ? g00 : g01;
      float gb = (c1 < 32) ? g10 : g11;
      float sv0 = (S0[r] + ga) * scale;
      float sv1 = (S1[r] + gb) * scale;
      float mloc = fmaxf(sv0, sv1);
#pragma unroll
      for (int msk = 1; msk < 16; msk <<= 1) mloc = fmaxf(mloc, __shfl_xor(mloc, msk, 32));
      float mnew = fmaxf(mrow[r], mloc);
      float corr = __expf(mrow[r] - mnew);
      mrow[r] = mnew;
      float p0 = __expf(sv0 - mnew);
      float p1 = __expf(sv1 - mnew);
      float rs = p0 + p1;
#pragma unroll
      for (int msk = 1; msk < 16; msk <<= 1) rs += __shfl_xor(rs, msk, 32);
      lrow[r] = lrow[r] * corr + rs;
      O0[r] *= corr; O1[r] *= corr; O2[r] *= corr; O3[r] *= corr;
      sl[i * 40 + nlo] = (bf16)p0;            // C-layout -> LDS (16x32 bf16)
      sl[i * 40 + nlo + 16] = (bf16)p1;
    }
    G0 = G2;                                  // carry leading band fragment
    asm volatile("s_wait_dscnt 0" ::: "memory");
    v16bf Pfrag = load_frag_a(sl + (size_t)nlo * 40, lane);   // A-layout reload
    // O += P (16x32) * V (32x64), vT rows contiguous in T
    O0 = wmma_bf16(Pfrag, load_frag_b(vth + (size_t)0  * TT + s0, TT, lane), O0);
    O1 = wmma_bf16(Pfrag, load_frag_b(vth + (size_t)16 * TT + s0, TT, lane), O1);
    O2 = wmma_bf16(Pfrag, load_frag_b(vth + (size_t)32 * TT + s0, TT, lane), O2);
    O3 = wmma_bf16(Pfrag, load_frag_b(vth + (size_t)48 * TT + s0, TT, lane), O3);
  }
  // normalize and store ctx (bf16, row-major [B*T, HK])
#pragma unroll
  for (int r = 0; r < 8; ++r) {
    float inv = 1.0f / lrow[r];
    O0[r] *= inv; O1[r] *= inv; O2[r] *= inv; O3[r] *= inv;
  }
  bf16* crow = ctx + (size_t)(b * TT + t0) * HK + h * KK;
#pragma unroll
  for (int r = 0; r < 8; ++r) {
    int i = r + 8 * hl;
    bf16* cp = crow + (size_t)i * HK;
    cp[nlo]      = (bf16)O0[r];
    cp[16 + nlo] = (bf16)O1[r];
    cp[32 + nlo] = (bf16)O2[r];
    cp[48 + nlo] = (bf16)O3[r];
  }
}

// ---------------- output projection + bias + residual (32x64 tiles) ----------------
__global__ __launch_bounds__(256, 1)
void out_proj(const bf16* __restrict__ ctx, const bf16* __restrict__ WoT,
              const float* __restrict__ bo, const float* __restrict__ x,
              float* __restrict__ out) {
  int lane = threadIdx.x & 31;
  int widx = blockIdx.x * 8 + (threadIdx.x >> 5);
  int mtile = widx >> 3, ntile = widx & 7;
  int n0 = ntile * 64;
  int m0 = mtile * 32;
  const bf16* arow0 = ctx + (size_t)(m0 + (lane & 15)) * HK;
  const bf16* arow1 = arow0 + (size_t)16 * HK;
  v8f acc[2][4] = {};
  for (int kc = 0; kc < HK; kc += 32) {
    v16bf a0 = load_frag_a(arow0 + kc, lane);
    v16bf a1 = load_frag_a(arow1 + kc, lane);
#pragma unroll
    for (int f = 0; f < 4; ++f) {
      v16bf bf = load_frag_b(WoT + (size_t)(n0 + f * 16) * HK + kc, HK, lane);
      acc[0][f] = wmma_bf16(a0, bf, acc[0][f]);
      acc[1][f] = wmma_bf16(a1, bf, acc[1][f]);
    }
  }
  int hl = lane >> 4, nlo = lane & 15;
#pragma unroll
  for (int ms = 0; ms < 2; ++ms) {
#pragma unroll
    for (int f = 0; f < 4; ++f) {
#pragma unroll
      for (int r = 0; r < 8; ++r) {
        size_t row = m0 + ms * 16 + r + 8 * hl;
        int col = n0 + f * 16 + nlo;
        out[row * DD + col] = acc[ms][f][r] + bo[col] + x[row * DD + col];
      }
    }
  }
}

// ---------------- host launcher ----------------
extern "C" void kernel_launch(void* const* d_in, const int* in_sizes, int n_in,
                              void* d_out, int out_size, void* d_ws, size_t ws_size,
                              hipStream_t stream) {
  const float* x    = (const float*)d_in[0];
  const float* pos  = (const float*)d_in[1];
  const float* ub   = (const float*)d_in[2];
  const float* vb   = (const float*)d_in[3];
  const float* lng  = (const float*)d_in[4];
  const float* lnb  = (const float*)d_in[5];
  const float* Wq   = (const float*)d_in[6];
  const float* bq   = (const float*)d_in[7];
  const float* Wk   = (const float*)d_in[8];
  const float* bk   = (const float*)d_in[9];
  const float* Wv   = (const float*)d_in[10];
  const float* bv   = (const float*)d_in[11];
  const float* Wp   = (const float*)d_in[12];
  const float* Wo   = (const float*)d_in[13];
  const float* bo   = (const float*)d_in[14];
  float* out = (float*)d_out;

  char* ws = (char*)d_ws;
  size_t off = 0;
  auto carve = [&](size_t bytes) -> void* {
    void* p = ws + off;
    off += (bytes + 255) & ~(size_t)255;
    return p;
  };
  bf16* ybuf = (bf16*)carve((size_t)BB * TT * DD * 2);   // 4 MB
  bf16* WqT  = (bf16*)carve((size_t)DD * HK * 2);
  bf16* WkT  = (bf16*)carve((size_t)DD * HK * 2);
  bf16* WvT  = (bf16*)carve((size_t)DD * HK * 2);
  bf16* WpT  = (bf16*)carve((size_t)DD * HK * 2);
  bf16* WoT  = (bf16*)carve((size_t)DD * HK * 2);
  bf16* posb = (bf16*)carve((size_t)BB * LL * DD * 2);   // 8 MB
  bf16* qu   = (bf16*)carve((size_t)BB * TT * HK * 2);
  bf16* qv   = (bf16*)carve((size_t)BB * TT * HK * 2);
  bf16* kbuf = (bf16*)carve((size_t)BB * TT * HK * 2);
  bf16* vTb  = (bf16*)carve((size_t)BB * HH * KK * TT * 2);
  bf16* pbuf = (bf16*)carve((size_t)BB * LP * HK * 2);   // 8 MB
  bf16* ctxb = (bf16*)carve((size_t)BB * TT * HK * 2);

  ln_kernel<<<BB * TT, 256, 0, stream>>>(x, lng, lnb, ybuf);
  prep_w<<<(512 * 512) / 256, 256, 0, stream>>>(Wq, Wk, Wv, Wp, Wo,
                                                WqT, WkT, WvT, WpT, WoT);
  int npos = BB * LL * DD;
  cvt_pos<<<(npos + 255) / 256, 256, 0, stream>>>(pos, posb, npos);

  // 32x64 wave tiles: q/k/v have 128 mtiles (128 blocks); p has 256 mtiles.
  gemm_proj<0><<<128, 256, 0, stream>>>(ybuf, WqT, bq, ub, vb, qu, qv);
  gemm_proj<1><<<128, 256, 0, stream>>>(ybuf, WkT, bk, nullptr, nullptr, kbuf, nullptr);
  gemm_proj<2><<<128, 256, 0, stream>>>(ybuf, WvT, bv, nullptr, nullptr, vTb, nullptr);
  gemm_proj<3><<<256, 256, 0, stream>>>(posb, WpT, nullptr, nullptr, nullptr, pbuf, nullptr);

  attn_kernel<<<BB * HH * (TT / 16) / 4, 128, 0, stream>>>(qu, qv, kbuf, pbuf, vTb, ctxb);
  out_proj<<<128, 256, 0, stream>>>(ctxb, WoT, bo, x, out);
}
